// LSTMTagger_65335042507515
// MI455X (gfx1250) — matrix operand order repaired and checked by reading, not compiled, not run on hardware
//
#include <hip/hip_runtime.h>
#include <stdint.h>

// ---------------- problem constants ----------------
#define VOCAB 200000
#define EMB   256
#define HID   512
#define G4    2048   // 4*HID
#define SEQ   8192
#define TOK   32768
#define TAGS  512
#define H2    1024   // 2*HID

typedef __attribute__((ext_vector_type(16))) __bf16 v16bf;
typedef __attribute__((ext_vector_type(8)))  float  v8f;

// ---------------- bf16 helpers (manual, RNE) ----------------
__device__ __forceinline__ unsigned short f2bf(float f) {
  unsigned int u = __float_as_uint(f);
  u += 0x7fffu + ((u >> 16) & 1u);   // round-to-nearest-even
  return (unsigned short)(u >> 16);
}
__device__ __forceinline__ float bf2f(unsigned short h) {
  return __uint_as_float(((unsigned int)h) << 16);
}

// ---------------- K0: f32 -> bf16 weight conversion ----------------
__global__ __launch_bounds__(256) void f32_to_bf16_k(
    const float* __restrict__ in, unsigned short* __restrict__ out, int n) {
  int i = blockIdx.x * 256 + threadIdx.x;
  if (i < n) out[i] = f2bf(in[i]);
}

// ---------------- K1: EmbeddingBag (gather + segment sum) ----------------
// segment_ids sorted -> binary search the token range per segment.
__global__ __launch_bounds__(256) void embed_bag_k(
    const float* __restrict__ emb, const int* __restrict__ ids,
    const int* __restrict__ seg, unsigned short* __restrict__ out) {
  int s = blockIdx.x;        // segment (morpheme) 0..SEQ-1
  int d = threadIdx.x;       // embedding dim 0..255
  int lo = 0, hi = TOK;
  while (lo < hi) { int m = (lo + hi) >> 1; if (seg[m] < s) lo = m + 1; else hi = m; }
  int st = lo;
  lo = st; hi = TOK;
  while (lo < hi) { int m = (lo + hi) >> 1; if (seg[m] < s + 1) lo = m + 1; else hi = m; }
  int en = lo;
  float acc = 0.f;
  for (int t = st; t < en; ++t)
    acc += emb[(size_t)ids[t] * EMB + d];
  out[(size_t)s * EMB + d] = f2bf(acc);
}

// ---------------- K2/K4: WMMA GEMM  C[M,N] = A[M,K](bf16) x B[N,K](bf16)^T + bias ----
// Block: 128 threads = 4 waves. Block tile 64(M) x 64(N); wave tile 16(M) x 64(N).
// A fragment (16x32 bf16): lane L: row = L%16, half = L/16;
//   pair p (uint32 = 2 bf16): k = (p/4)*16 + half*8 + (p%4)*2   (ISA 7.12.2)
// B fragment (32x16 bf16): lane L: col = L%16, grp = L/16;
//   pair v: k = (v/4)*16 + grp*8 + (v%4)*2
// D: VGPR i -> row = half*8 + i, col = L%16
__global__ __launch_bounds__(128) void gemm_bf16_bias_k(
    const unsigned short* __restrict__ A, const unsigned short* __restrict__ B,
    const float* __restrict__ bias, float* __restrict__ C,
    int M, int N, int K) {
  int tid  = threadIdx.x;
  int wv   = tid >> 5;
  int lane = tid & 31;
  int half = lane >> 4;
  int l16  = lane & 15;
  int Mbase = blockIdx.x * 64 + wv * 16;
  int Nbase = blockIdx.y * 64;

  v8f acc0 = {}, acc1 = {}, acc2 = {}, acc3 = {};
  v8f* accs[4] = {&acc0, &acc1, &acc2, &acc3};

  const unsigned short* arow = A + (size_t)(Mbase + l16) * K;

  for (int kb = 0; kb < K; kb += 32) {
    union { unsigned int u[8]; v16bf v; } af;
#pragma unroll
    for (int p = 0; p < 8; ++p) {
      int kk = ((p >> 2) << 4) + half * 8 + ((p & 3) << 1);
      af.u[p] = *(const unsigned int*)(arow + kb + kk);
    }
#pragma unroll
    for (int nt = 0; nt < 4; ++nt) {
      int ncol = Nbase + nt * 16 + l16;
      const unsigned short* brow = B + (size_t)ncol * K + kb;
      union { unsigned int u[8]; v16bf v; } bfr;
#pragma unroll
      for (int v = 0; v < 8; ++v) {
        int kk = ((v >> 2) << 4) + half * 8 + ((v & 3) << 1);
        bfr.u[v] = *(const unsigned int*)(brow + kk);
      }
      *accs[nt] = __builtin_amdgcn_wmma_f32_16x16x32_bf16(
          false, af.v, false, bfr.v, (short)0, *accs[nt], false, false);
    }
  }
#pragma unroll
  for (int nt = 0; nt < 4; ++nt) {
    int ncol = Nbase + nt * 16 + l16;
    float bv = bias[ncol];
    v8f a = *accs[nt];
#pragma unroll
    for (int i = 0; i < 8; ++i) {
      int mrow = Mbase + half * 8 + i;
      C[(size_t)mrow * N + ncol] = a[i] + bv;
    }
  }
}

// ---------------- K3: bidirectional LSTM recurrence ----------------
// 32 workgroups x 512 threads. dir = blk/16, slice w = blk%16 owns hidden
// j in [w*32, w*32+32) -> 128 rows of W_hh (4 gates x 32), resident in LDS
// (128 KB bf16). h(t) exchanged through L2 (double-buffered), per-step sync =
// CDNA5 cluster barrier (NOP if not cluster-launched) + software barrier.
__device__ __forceinline__ void dir_barrier(int* bar, int tid) {
  __threadfence();          // make this WG's h writes agent-visible
  __syncthreads();
  if (tid == 0) {
    __builtin_amdgcn_s_cluster_barrier();   // s_barrier_signal/-wait -3 (NOP outside cluster)
    int gen  = __hip_atomic_load(bar + 1, __ATOMIC_RELAXED, __HIP_MEMORY_SCOPE_AGENT);
    int prev = __hip_atomic_fetch_add(bar, 1, __ATOMIC_ACQ_REL, __HIP_MEMORY_SCOPE_AGENT);
    if (prev == 15) {
      __hip_atomic_store(bar, 0, __ATOMIC_RELAXED, __HIP_MEMORY_SCOPE_AGENT);
      __hip_atomic_fetch_add(bar + 1, 1, __ATOMIC_RELEASE, __HIP_MEMORY_SCOPE_AGENT);
    } else {
      while (__hip_atomic_load(bar + 1, __ATOMIC_ACQUIRE, __HIP_MEMORY_SCOPE_AGENT) == gen)
        __builtin_amdgcn_s_sleep(4);
    }
  }
  __syncthreads();
}

__device__ __forceinline__ float sigmoidf_(float x) { return 1.f / (1.f + expf(-x)); }

__global__ __launch_bounds__(512) void lstm_bidir_k(
    const float* __restrict__ xg_f, const float* __restrict__ xg_b,
    const unsigned short* __restrict__ whh_f, const unsigned short* __restrict__ whh_b,
    unsigned short* __restrict__ hbuf,   // [2 dirs][2 bufs][HID] bf16
    unsigned short* __restrict__ Hcat,   // [SEQ][H2] bf16
    int* __restrict__ barr) {            // [2 dirs][32 ints]
  int blk = blockIdx.x;
  int dir = blk >> 4;
  int w   = blk & 15;
  int tid = threadIdx.x;

  const float*          xg  = dir ? xg_b  : xg_f;
  const unsigned short* whh = dir ? whh_b : whh_f;
  unsigned short* hb = hbuf + (size_t)dir * 2 * HID;
  int* bar = barr + dir * 32;
  int j0 = w * 32;

  __shared__ unsigned short wlds[128 * HID];   // 128 KB
  __shared__ float h_s[HID];
  __shared__ float part[512];
  __shared__ float gv[128];
  __shared__ float c_s[32];

  // stage this WG's W_hh slice into LDS: row r = g*32 + jj  <-  W_hh[g*512 + j0+jj]
  for (int i = tid; i < 128 * HID; i += 512) {
    int r = i >> 9, k = i & (HID - 1);
    int g = r >> 5, jj = r & 31;
    wlds[i] = whh[((size_t)(g * HID + j0 + jj)) * HID + k];
  }
  if (tid < 32) c_s[tid] = 0.f;
  __syncthreads();

  int r = tid >> 2;          // output row 0..127 owned by this thread group
  int p = tid & 3;           // k-partition 0..3 (128 k's each)

  for (int t = 0; t < SEQ; ++t) {
    // read h(t-1) from the other buffer (both zeroed before launch)
    const unsigned short* hp = hb + (((t & 1) ^ 1) * HID);
    h_s[tid] = bf2f(hp[tid]);
    __syncthreads();

    const unsigned int* wrow = (const unsigned int*)(wlds + (size_t)r * HID + p * 128);
    const float2*       hrow = (const float2*)(h_s + p * 128);
    float acc = 0.f;
#pragma unroll 8
    for (int q = 0; q < 64; ++q) {
      unsigned int wvp = wrow[q];
      float2 hv = hrow[q];
      acc = fmaf(bf2f((unsigned short)(wvp & 0xffffu)), hv.x, acc);
      acc = fmaf(bf2f((unsigned short)(wvp >> 16)),     hv.y, acc);
    }
    part[tid] = acc;
    __syncthreads();

    int trow = dir ? (SEQ - 1 - t) : t;
    if (tid < 128) {
      int g = tid >> 5, jj = tid & 31;
      int n = g * HID + j0 + jj;
      gv[tid] = xg[(size_t)trow * G4 + n] +
                part[tid * 4] + part[tid * 4 + 1] + part[tid * 4 + 2] + part[tid * 4 + 3];
    }
    __syncthreads();

    if (tid < 32) {
      float iv = sigmoidf_(gv[tid]);
      float fv = sigmoidf_(gv[32 + tid]);
      float gg = tanhf(gv[64 + tid]);
      float ov = sigmoidf_(gv[96 + tid]);
      float c  = fv * c_s[tid] + iv * gg;
      c_s[tid] = c;
      unsigned short hq = f2bf(ov * tanhf(c));
      hb[(t & 1) * HID + j0 + tid] = hq;                         // for next step
      Hcat[(size_t)trow * H2 + dir * HID + j0 + tid] = hq;       // for output GEMM
    }
    dir_barrier(bar, tid);
  }
}

// ---------------- K5: row-wise log-softmax over TAGS=512 ----------------
__global__ __launch_bounds__(256) void log_softmax_k(float* __restrict__ x) {
  __shared__ float red[256];
  int row = blockIdx.x, tid = threadIdx.x;
  float* p = x + (size_t)row * TAGS;
  float a = p[tid], b = p[tid + 256];
  red[tid] = fmaxf(a, b);
  __syncthreads();
  for (int s = 128; s > 0; s >>= 1) {
    if (tid < s) red[tid] = fmaxf(red[tid], red[tid + s]);
    __syncthreads();
  }
  float mx = red[0];
  __syncthreads();
  red[tid] = expf(a - mx) + expf(b - mx);
  __syncthreads();
  for (int s = 128; s > 0; s >>= 1) {
    if (tid < s) red[tid] += red[tid + s];
    __syncthreads();
  }
  float lse = mx + logf(red[0]);
  p[tid] = a - lse;
  p[tid + 256] = b - lse;
}

// ---------------- host: workspace layout + launch ----------------
static inline size_t align256(size_t x) { return (x + 255) & ~(size_t)255; }

extern "C" void kernel_launch(void* const* d_in, const int* in_sizes, int n_in,
                              void* d_out, int out_size, void* d_ws, size_t ws_size,
                              hipStream_t stream) {
  (void)in_sizes; (void)n_in; (void)out_size; (void)ws_size;
  const float* emb    = (const float*)d_in[0];
  const float* w_ih_f = (const float*)d_in[1];
  const float* w_hh_f = (const float*)d_in[2];
  const float* b_f    = (const float*)d_in[3];
  const float* w_ih_b = (const float*)d_in[4];
  const float* w_hh_b = (const float*)d_in[5];
  const float* b_b    = (const float*)d_in[6];
  const float* w_out  = (const float*)d_in[7];
  const float* b_out  = (const float*)d_in[8];
  const int*   ids    = (const int*)d_in[9];
  const int*   seg    = (const int*)d_in[10];

  char* ws = (char*)d_ws;
  size_t o = 0;
  unsigned short* embeds  = (unsigned short*)(ws + o); o = align256(o + (size_t)SEQ * EMB * 2);
  unsigned short* wihf_bf = (unsigned short*)(ws + o); o = align256(o + (size_t)G4 * EMB * 2);
  unsigned short* wihb_bf = (unsigned short*)(ws + o); o = align256(o + (size_t)G4 * EMB * 2);
  unsigned short* whhf_bf = (unsigned short*)(ws + o); o = align256(o + (size_t)G4 * HID * 2);
  unsigned short* whhb_bf = (unsigned short*)(ws + o); o = align256(o + (size_t)G4 * HID * 2);
  unsigned short* wout_bf = (unsigned short*)(ws + o); o = align256(o + (size_t)TAGS * H2 * 2);
  float*          xg_f    = (float*)(ws + o);          o = align256(o + (size_t)SEQ * G4 * 4);
  float*          xg_b    = (float*)(ws + o);          o = align256(o + (size_t)SEQ * G4 * 4);
  unsigned short* Hcat    = (unsigned short*)(ws + o); o = align256(o + (size_t)SEQ * H2 * 2);
  unsigned short* hbuf    = (unsigned short*)(ws + o); o = align256(o + (size_t)2 * 2 * HID * 2);
  int*            barr    = (int*)(ws + o);            o = align256(o + 256);

  // K0: weight conversions to bf16
  f32_to_bf16_k<<<(G4 * EMB + 255) / 256, 256, 0, stream>>>(w_ih_f, wihf_bf, G4 * EMB);
  f32_to_bf16_k<<<(G4 * EMB + 255) / 256, 256, 0, stream>>>(w_ih_b, wihb_bf, G4 * EMB);
  f32_to_bf16_k<<<(G4 * HID + 255) / 256, 256, 0, stream>>>(w_hh_f, whhf_bf, G4 * HID);
  f32_to_bf16_k<<<(G4 * HID + 255) / 256, 256, 0, stream>>>(w_hh_b, whhb_bf, G4 * HID);
  f32_to_bf16_k<<<(TAGS * H2 + 255) / 256, 256, 0, stream>>>(w_out, wout_bf, TAGS * H2);

  // K1: embedding bag
  embed_bag_k<<<SEQ, 256, 0, stream>>>(emb, ids, seg, embeds);

  // K2: xg = embeds @ w_ih^T + b   (WMMA, M=8192 N=2048 K=256)
  dim3 gx(SEQ / 64, G4 / 64);
  gemm_bf16_bias_k<<<gx, 128, 0, stream>>>(embeds, wihf_bf, b_f, xg_f, SEQ, G4, EMB);
  gemm_bf16_bias_k<<<gx, 128, 0, stream>>>(embeds, wihb_bf, b_b, xg_b, SEQ, G4, EMB);

  // K3: recurrence (zero h double-buffers + barrier state each launch)
  hipMemsetAsync(hbuf, 0, (size_t)2 * 2 * HID * 2, stream);
  hipMemsetAsync(barr, 0, 256, stream);
  lstm_bidir_k<<<32, 512, 0, stream>>>(xg_f, xg_b, whhf_bf, whhb_bf, hbuf, Hcat, barr);

  // K4: logits = Hcat @ w_out^T + b_out  (WMMA, M=8192 N=512 K=1024)
  dim3 gl(SEQ / 64, TAGS / 64);
  gemm_bf16_bias_k<<<gl, 128, 0, stream>>>(Hcat, wout_bf, b_out, (float*)d_out, SEQ, TAGS, H2);

  // K5: in-place log-softmax
  log_softmax_k<<<SEQ, 256, 0, stream>>>((float*)d_out);
}
